// MultiheadAttentionQuant_90048284328336
// MI455X (gfx1250) — matrix operand order repaired and checked by reading, not compiled, not run on hardware
//
#include <hip/hip_runtime.h>
#include <hip/hip_bf16.h>

// CDNA5 / gfx1250 wave32 WMMA implementation of quantized multi-head attention.
// All five GEMM stages use v_wmma_f32_16x16x32_f16; the dense projections stage
// their A panels through LDS with GLOBAL_LOAD_ASYNC_TO_LDS_B128 (double
// buffered, s_wait_asynccnt) when the toolchain exposes the builtin.

typedef __attribute__((ext_vector_type(16))) _Float16 v16h;
typedef __attribute__((ext_vector_type(8)))  _Float16 v8h;
typedef __attribute__((ext_vector_type(8)))  float    v8f;
typedef __attribute__((ext_vector_type(4)))  int      v4i;

#define WMMA_F16(a, b, c) \
  __builtin_amdgcn_wmma_f32_16x16x32_f16(false, (a), false, (b), (short)0, (c), false, false)

#define FINF __builtin_inff()

#if defined(__HIP_DEVICE_COMPILE__) && \
    __has_builtin(__builtin_amdgcn_global_load_async_to_lds_b128) && \
    __has_builtin(__builtin_amdgcn_s_wait_asynccnt)
#define USE_ASYNC_LDS 1
typedef __attribute__((address_space(1))) void gvoid;   // global (prints as __device__)
typedef __attribute__((address_space(3))) void lvoid;   // LDS
typedef __attribute__((address_space(1))) v4i  gv4i;
typedef __attribute__((address_space(3))) v4i  lv4i;
#endif

// ---------------------------------------------------------------------------
// float atomics via CAS loop
// ---------------------------------------------------------------------------
__device__ __forceinline__ void atomic_max_f(float* addr, float val) {
  unsigned int* ai = (unsigned int*)addr;
  unsigned int old = __float_as_uint(*addr);
  while (val > __uint_as_float(old)) {
    unsigned int assumed = old;
    old = atomicCAS(ai, assumed, __float_as_uint(val));
    if (old == assumed) break;
  }
}
__device__ __forceinline__ void atomic_min_f(float* addr, float val) {
  unsigned int* ai = (unsigned int*)addr;
  unsigned int old = __float_as_uint(*addr);
  while (val < __uint_as_float(old)) {
    unsigned int assumed = old;
    old = atomicCAS(ai, assumed, __float_as_uint(val));
    if (old == assumed) break;
  }
}

// fake_quant: q = rint(trunc(clamp(x,mn,mx)/s)*s + mn), s = (mx-mn)/255
__device__ __forceinline__ float fq(float x, float mn, float mx) {
  float s = fmaxf((mx - mn) * (1.0f / 255.0f), 1e-20f);
  float c = fminf(fmaxf(x, mn), mx);
  return rintf(truncf(c / s) * s + mn);
}

// ---------------------------------------------------------------------------
// WMMA tile loaders (16-bit, wave32; contiguous k = 16*(lane/16) + s, same order in A and B)
// A 16x32 (MxK): lane<16 -> M=lane, halves[0:16]=K0..15
//                lane>=16 -> M=lane-16, K16..31
// ---------------------------------------------------------------------------
__device__ __forceinline__ v16h load_a_tile(const _Float16* base, int ldk, int lane) {
  int m = lane & 15;
  int koff = (lane < 16) ? 0 : 16;
  const _Float16* p = base + (size_t)m * (size_t)ldk;
  v8h lo = *(const v8h*)(p + koff);
  v8h hi = *(const v8h*)(p + 8 + koff);
  v16h a;
#pragma unroll
  for (int i = 0; i < 8; ++i) { a[i] = lo[i]; a[i + 8] = hi[i]; }
  return a;
}

// B 32x16 (KxN): lane = N (mod 16); lane<16 holds K0..15, lane>=16 holds K16..31
__device__ __forceinline__ v16h load_b_tile(const _Float16* base, int ldk, int lane) {
  int n = lane & 15;
  int koff = (lane < 16) ? 0 : 16;
  return *(const v16h*)(base + (size_t)n * (size_t)ldk + koff);
}

// ---------------------------------------------------------------------------
// stats init: pairs [min,max] -> [+inf,-inf]
// ---------------------------------------------------------------------------
__global__ void init_stats_kernel(float* stats, int n) {
  int i = blockIdx.x * blockDim.x + threadIdx.x;
  if (i < n) stats[i] = (i & 1) ? -FINF : FINF;
}

// global min/max of a tensor -> pair
__global__ void minmax_kernel(const float* __restrict__ x, size_t n, float* pair) {
  __shared__ float smn[256], smx[256];
  float mn = FINF, mx = -FINF;
  for (size_t i = (size_t)blockIdx.x * blockDim.x + threadIdx.x; i < n;
       i += (size_t)gridDim.x * blockDim.x) {
    float v = x[i];
    mn = fminf(mn, v);
    mx = fmaxf(mx, v);
  }
  smn[threadIdx.x] = mn; smx[threadIdx.x] = mx;
  __syncthreads();
  for (int s = 128; s > 0; s >>= 1) {
    if ((int)threadIdx.x < s) {
      smn[threadIdx.x] = fminf(smn[threadIdx.x], smn[threadIdx.x + s]);
      smx[threadIdx.x] = fmaxf(smx[threadIdx.x], smx[threadIdx.x + s]);
    }
    __syncthreads();
  }
  if (threadIdx.x == 0) {
    atomic_min_f(&pair[0], smn[0]);
    atomic_max_f(&pair[1], smx[0]);
  }
}

// global-stat fake-quant, f32 -> f16
__global__ void quant_global_kernel(const float* __restrict__ x, size_t n,
                                    const float* pair, _Float16* __restrict__ out) {
  size_t i = (size_t)blockIdx.x * blockDim.x + threadIdx.x;
  if (i >= n) return;
  out[i] = (_Float16)fq(x[i], pair[0], pair[1]);
}

// f32 -> f16 weight conversion
__global__ void cvt_f16_kernel(const float* __restrict__ x, size_t n, _Float16* __restrict__ out) {
  size_t i = (size_t)blockIdx.x * blockDim.x + threadIdx.x;
  if (i >= n) return;
  out[i] = (_Float16)x[i];
}

// per-head fake-quant; optional transpose to (B,H,dk,S) layout (for V)
__global__ void quant_perhead_kernel(const float* __restrict__ x, const float* pairs,
                                     _Float16* __restrict__ out, int transpose,
                                     int B, int S, int D, int H) {
  size_t n = (size_t)B * S * D;
  size_t i = (size_t)blockIdx.x * blockDim.x + threadIdx.x;
  if (i >= n) return;
  int col = (int)(i % D);
  size_t row = i / D;
  int h = col >> 6;                       // dk = 64
  _Float16 q = (_Float16)fq(x[i], pairs[2 * h], pairs[2 * h + 1]);
  if (!transpose) {
    out[i] = q;
  } else {
    int b = (int)(row / S), s = (int)(row % S), dl = col & 63;
    out[((size_t)(b * H + h) * 64 + dl) * (size_t)S + s] = q;
  }
}

// ---------------------------------------------------------------------------
// Dense GEMM: C[M,N] = A16[M,K] @ W16[N,K]^T + bias (+ per-head min/max).
// Block = 8 waves sharing one 16-row A strip staged in LDS (async double
// buffer); each wave owns a 16x64 C tile (= exactly one head), reusing the A
// fragment across 4 WMMAs per 32-wide k-chunk.
// ---------------------------------------------------------------------------
__global__ void gemm_wmma_kernel(const _Float16* __restrict__ A, const _Float16* __restrict__ W,
                                 const float* __restrict__ bias, float* __restrict__ C,
                                 float* headStats, int M, int N, int K) {
  __shared__ __align__(32) _Float16 sA[2][16][40];  // 80B row stride: aligned, conflict-spread

  const int tid = (int)threadIdx.x;
  const int lane = tid & 31;
  const int wid = tid >> 5;
  const int nblk = N >> 9;                 // 512 cols per block
  const int tm = blockIdx.x / nblk;        // 16-row strip
  const int bn = blockIdx.x % nblk;
  const int col0 = bn * 512 + wid * 64;    // this wave's 64-col (one-head) slab
  if (tm * 16 >= M) return;

  const _Float16* astrip = A + (size_t)(tm * 16) * (size_t)K;
  const int row = tid >> 2, seg = tid & 3; // staging assignment for tid < 64

  // stage one 16x32 A chunk into sA[buf] (threads 0..63, 16B each)
#if defined(USE_ASYNC_LDS)
#define STAGE_A(buf, kc)                                                              \
  do {                                                                                \
    if (tid < 64) {                                                                   \
      const _Float16* gp = astrip + (size_t)row * (size_t)K + (kc) + seg * 8;         \
      __builtin_amdgcn_global_load_async_to_lds_b128(                                 \
          (gv4i*)(gvoid*)gp, (lv4i*)(lvoid*)&sA[buf][row][seg * 8], 0, 0);            \
    }                                                                                 \
  } while (0)
#define WAIT_A() __builtin_amdgcn_s_wait_asynccnt(0)
#else
#define STAGE_A(buf, kc)                                                              \
  do {                                                                                \
    if (tid < 64) {                                                                   \
      const _Float16* gp = astrip + (size_t)row * (size_t)K + (kc) + seg * 8;         \
      *(v8h*)&sA[buf][row][seg * 8] = *(const v8h*)gp;                                \
    }                                                                                 \
  } while (0)
#define WAIT_A() ((void)0)
#endif

  STAGE_A(0, 0);
  WAIT_A();
  __syncthreads();

  v8f zero = {};
  v8f acc[4] = {zero, zero, zero, zero};
  const int am = lane & 15;
  const int akoff = (lane < 16) ? 0 : 16;

  for (int kc = 0; kc < K; kc += 32) {
    const int par = (kc >> 5) & 1;
    if (kc + 32 < K) STAGE_A(1 - par, kc + 32);   // overlap next A panel with compute

    // A fragment from LDS (ds_load_b128 x2)
    v8h lo = *(const v8h*)&sA[par][am][akoff];
    v8h hi = *(const v8h*)&sA[par][am][8 + akoff];
    v16h a;
#pragma unroll
    for (int i = 0; i < 8; ++i) { a[i] = lo[i]; a[i + 8] = hi[i]; }

#pragma unroll
    for (int nt = 0; nt < 4; ++nt) {
      const _Float16* wbase = W + (size_t)(col0 + nt * 16) * (size_t)K + kc;
      __builtin_prefetch(wbase + 256, 0, 0);
      acc[nt] = WMMA_F16(a, load_b_tile(wbase, K, lane), acc[nt]);
    }

    WAIT_A();
    __syncthreads();
  }

  const int n = lane & 15, hl = lane >> 4;
  float lmn = FINF, lmx = -FINF;
#pragma unroll
  for (int nt = 0; nt < 4; ++nt) {
    float bval = bias[col0 + nt * 16 + n];
#pragma unroll
    for (int r = 0; r < 8; ++r) {
      float v = acc[nt][r] + bval;
      int orow = tm * 16 + r + 8 * hl;
      C[(size_t)orow * (size_t)N + col0 + nt * 16 + n] = v;
      lmn = fminf(lmn, v);
      lmx = fmaxf(lmx, v);
    }
  }
  if (headStats) {
#pragma unroll
    for (int off = 16; off > 0; off >>= 1) {
      lmn = fminf(lmn, __shfl_xor(lmn, off, 32));
      lmx = fmaxf(lmx, __shfl_xor(lmx, off, 32));
    }
    if (lane == 0) {
      int head = col0 >> 6;               // one head per wave
      atomic_min_f(&headStats[2 * head], lmn);
      atomic_max_f(&headStats[2 * head + 1], lmx);
    }
  }
#undef STAGE_A
#undef WAIT_A
}

// ---------------------------------------------------------------------------
// Attention pass A: scores strip (16 x S) via WMMA; row max m, exp-sum l,
// row-min; per-head p stats: pmax = max(1/l), pmin = min(exp(smin-m)/l).
// One 32-thread block per (b, h, 16-row strip).
// ---------------------------------------------------------------------------
__global__ void attn_pass_a_kernel(const _Float16* __restrict__ Q, const _Float16* __restrict__ Kh,
                                   float* __restrict__ mbuf, float* __restrict__ lbuf,
                                   float* pstats, int B, int S, int D, int H) {
  __shared__ float sc[16][17];
  int lane = (int)threadIdx.x;
  int spb = S >> 4;
  int b = blockIdx.x / (H * spb);
  int rem = blockIdx.x % (H * spb);
  int h = rem / spb, strip = rem % spb;

  const _Float16* qbase = Q + ((size_t)b * S + strip * 16) * (size_t)D + h * 64;
  v16h a0 = load_a_tile(qbase, D, lane);
  v16h a1 = load_a_tile(qbase + 32, D, lane);

  int n = lane & 15, hl = lane >> 4;
  float m = -FINF, rmin = FINF, lsum = 0.0f;

  for (int kt = 0; kt < (S >> 4); ++kt) {
    const _Float16* kbase = Kh + ((size_t)b * S + kt * 16) * (size_t)D + h * 64;
    v8f c = {};
    c = WMMA_F16(a0, load_b_tile(kbase, D, lane), c);
    c = WMMA_F16(a1, load_b_tile(kbase + 32, D, lane), c);
#pragma unroll
    for (int r = 0; r < 8; ++r) sc[r + 8 * hl][n] = c[r] * 0.125f;  // 1/sqrt(64)
    __syncthreads();
    if (lane < 16) {
      float tmax = -FINF;
#pragma unroll
      for (int j = 0; j < 16; ++j) {
        float s = sc[lane][j];
        tmax = fmaxf(tmax, s);
        rmin = fminf(rmin, s);
      }
      float nm = fmaxf(m, tmax);
      lsum = lsum * __expf(m - nm);
#pragma unroll
      for (int j = 0; j < 16; ++j) lsum += __expf(sc[lane][j] - nm);
      m = nm;
    }
    __syncthreads();
  }
  if (lane < 16) {
    size_t ridx = ((size_t)(b * H + h)) * (size_t)S + strip * 16 + lane;
    mbuf[ridx] = m;
    lbuf[ridx] = lsum;
    atomic_max_f(&pstats[2 * h + 1], 1.0f / lsum);
    atomic_min_f(&pstats[2 * h + 0], __expf(rmin - m) / lsum);
  }
}

// ---------------------------------------------------------------------------
// Attention pass B: recompute scores (WMMA), p = fq(softmax), transpose p
// tile through LDS into A-layout, WMMA against V^T -> x strip; fused global
// x min/max atomics.
// ---------------------------------------------------------------------------
__global__ void attn_pass_b_kernel(const _Float16* __restrict__ Q, const _Float16* __restrict__ Kh,
                                   const _Float16* __restrict__ Vt,
                                   const float* __restrict__ mbuf, const float* __restrict__ lbuf,
                                   const float* pstats, float* __restrict__ xf, float* xstats,
                                   int B, int S, int D, int H) {
  __shared__ __align__(32) _Float16 pl[16][40];  // 80B row stride: keeps v8h reads aligned
  __shared__ float mrow[16], lrow[16];
  int lane = (int)threadIdx.x;
  int spb = S >> 4;
  int b = blockIdx.x / (H * spb);
  int rem = blockIdx.x % (H * spb);
  int h = rem / spb, strip = rem % spb;

  if (lane < 16) {
    size_t ridx = ((size_t)(b * H + h)) * (size_t)S + strip * 16 + lane;
    mrow[lane] = mbuf[ridx];
    lrow[lane] = lbuf[ridx];
  }
  __syncthreads();

  float qmn = pstats[2 * h], qmx = pstats[2 * h + 1];
  const _Float16* qbase = Q + ((size_t)b * S + strip * 16) * (size_t)D + h * 64;
  v16h a0 = load_a_tile(qbase, D, lane);
  v16h a1 = load_a_tile(qbase + 32, D, lane);

  v8f zero = {};
  v8f o[4] = {zero, zero, zero, zero};
  int n = lane & 15, hl = lane >> 4;
  const _Float16* vrowbase = Vt + ((size_t)(b * H + h) * 64) * (size_t)S;

  for (int kc = 0; kc < S; kc += 32) {
#pragma unroll
    for (int t = 0; t < 2; ++t) {
      const _Float16* kbase = Kh + ((size_t)b * S + kc + t * 16) * (size_t)D + h * 64;
      v8f c = {};
      c = WMMA_F16(a0, load_b_tile(kbase, D, lane), c);
      c = WMMA_F16(a1, load_b_tile(kbase + 32, D, lane), c);
#pragma unroll
      for (int r = 0; r < 8; ++r) {
        int row = r + 8 * hl;
        float s = c[r] * 0.125f;
        float p = __expf(s - mrow[row]) / lrow[row];
        pl[row][t * 16 + n] = (_Float16)fq(p, qmn, qmx);
      }
    }
    __syncthreads();
    // rebuild p tile in A layout
    int mr = lane & 15, koff = (lane < 16) ? 0 : 16;
    v8h lo = *(const v8h*)&pl[mr][koff];
    v8h hi = *(const v8h*)&pl[mr][8 + koff];
    v16h ap;
#pragma unroll
    for (int i = 0; i < 8; ++i) { ap[i] = lo[i]; ap[i + 8] = hi[i]; }
    __syncthreads();

    size_t kvo = (size_t)kc + ((lane < 16) ? 0 : 16);
#pragma unroll
    for (int nt = 0; nt < 4; ++nt) {
      v16h bv = *(const v16h*)(vrowbase + (size_t)(nt * 16 + n) * (size_t)S + kvo);
      o[nt] = WMMA_F16(ap, bv, o[nt]);
    }
  }

  float lmn = FINF, lmx = -FINF;
#pragma unroll
  for (int nt = 0; nt < 4; ++nt) {
#pragma unroll
    for (int r = 0; r < 8; ++r) {
      float v = o[nt][r];
      int grow = b * S + strip * 16 + r + 8 * hl;
      xf[(size_t)grow * (size_t)D + h * 64 + nt * 16 + n] = v;
      lmn = fminf(lmn, v);
      lmx = fmaxf(lmx, v);
    }
  }
#pragma unroll
  for (int off = 16; off > 0; off >>= 1) {
    lmn = fminf(lmn, __shfl_xor(lmn, off, 32));
    lmx = fmaxf(lmx, __shfl_xor(lmx, off, 32));
  }
  if (lane == 0) {
    atomic_min_f(&xstats[0], lmn);
    atomic_max_f(&xstats[1], lmx);
  }
}

// ---------------------------------------------------------------------------
extern "C" void kernel_launch(void* const* d_in, const int* in_sizes, int n_in,
                              void* d_out, int out_size, void* d_ws, size_t ws_size,
                              hipStream_t stream) {
  const float* query = (const float*)d_in[0];
  const float* key   = (const float*)d_in[1];
  const float* value = (const float*)d_in[2];
  const float* Wq = (const float*)d_in[3]; const float* bq = (const float*)d_in[4];
  const float* Wk = (const float*)d_in[5]; const float* bk = (const float*)d_in[6];
  const float* Wv = (const float*)d_in[7]; const float* bv = (const float*)d_in[8];
  const float* Wo = (const float*)d_in[9]; const float* bo = (const float*)d_in[10];
  float* out = (float*)d_out;

  const int B = 2, S = 1024, D = 1024, H = 16;
  const size_t NEL = (size_t)B * S * D;  // 2M
  const size_t WEL = (size_t)D * D;      // 1M

  // workspace carve-up (~56.3 MB total)
  char* ws = (char*)d_ws;
  size_t off = 0;
  auto take = [&](size_t bytes) -> char* {
    char* p = ws + off;
    off = (off + bytes + 255) & ~(size_t)255;
    return p;
  };
  float*    stats = (float*)take(256 * sizeof(float));
  _Float16* q16   = (_Float16*)take(NEL * 2);
  _Float16* k16   = (_Float16*)take(NEL * 2);
  _Float16* v16   = (_Float16*)take(NEL * 2);
  _Float16* Wq16  = (_Float16*)take(WEL * 2);
  _Float16* Wk16  = (_Float16*)take(WEL * 2);
  _Float16* Wv16  = (_Float16*)take(WEL * 2);
  _Float16* Wo16  = (_Float16*)take(WEL * 2);
  float*    Qf    = (float*)take(NEL * 4);
  float*    Kf    = (float*)take(NEL * 4);
  float*    Vf    = (float*)take(NEL * 4);
  _Float16* Qh16  = (_Float16*)take(NEL * 2);
  _Float16* Kh16  = (_Float16*)take(NEL * 2);
  _Float16* Vt16  = (_Float16*)take(NEL * 2);
  float*    mbuf  = (float*)take((size_t)B * H * S * 4);
  float*    lbuf  = (float*)take((size_t)B * H * S * 4);
  float*    xf    = Qf;              // alias: Qf dead after Qh16 is built
  _Float16* x16   = (_Float16*)Kf;   // alias: Kf dead after Kh16 is built

  // stat pairs: [min,max] each
  float* stQ  = stats + 0;
  float* stK  = stats + 2;
  float* stV  = stats + 4;
  float* stQH = stats + 6;         // 16 pairs
  float* stKH = stats + 38;        // 16 pairs
  float* stVH = stats + 70;        // 16 pairs
  float* stP  = stats + 102;       // 16 pairs
  float* stX  = stats + 134;       // 1 pair -> 136 floats total

  const int EW = (int)((NEL + 255) / 256);
  const int WW = (int)((WEL + 255) / 256);
  const int GEMM_BLOCKS = ((B * S) / 16) * (D / 512);  // 16-row strip x 512-col slab
  const int ATTN_BLOCKS = B * H * (S / 16);

  init_stats_kernel<<<1, 256, 0, stream>>>(stats, 136);

  // 1) global input stats + quant to f16
  minmax_kernel<<<1024, 256, 0, stream>>>(query, NEL, stQ);
  minmax_kernel<<<1024, 256, 0, stream>>>(key,   NEL, stK);
  minmax_kernel<<<1024, 256, 0, stream>>>(value, NEL, stV);
  quant_global_kernel<<<EW, 256, 0, stream>>>(query, NEL, stQ, q16);
  quant_global_kernel<<<EW, 256, 0, stream>>>(key,   NEL, stK, k16);
  quant_global_kernel<<<EW, 256, 0, stream>>>(value, NEL, stV, v16);

  // 2) weights to f16
  cvt_f16_kernel<<<WW, 256, 0, stream>>>(Wq, WEL, Wq16);
  cvt_f16_kernel<<<WW, 256, 0, stream>>>(Wk, WEL, Wk16);
  cvt_f16_kernel<<<WW, 256, 0, stream>>>(Wv, WEL, Wv16);
  cvt_f16_kernel<<<WW, 256, 0, stream>>>(Wo, WEL, Wo16);

  // 3) projections (WMMA, async LDS A-staging) with fused per-head min/max
  gemm_wmma_kernel<<<GEMM_BLOCKS, 256, 0, stream>>>(q16, Wq16, bq, Qf, stQH, B * S, D, D);
  gemm_wmma_kernel<<<GEMM_BLOCKS, 256, 0, stream>>>(k16, Wk16, bk, Kf, stKH, B * S, D, D);
  gemm_wmma_kernel<<<GEMM_BLOCKS, 256, 0, stream>>>(v16, Wv16, bv, Vf, stVH, B * S, D, D);

  // 4) per-head quant to f16 (V also transposed to (B,H,dk,S))
  quant_perhead_kernel<<<EW, 256, 0, stream>>>(Qf, stQH, Qh16, 0, B, S, D, H);
  quant_perhead_kernel<<<EW, 256, 0, stream>>>(Kf, stKH, Kh16, 0, B, S, D, H);
  quant_perhead_kernel<<<EW, 256, 0, stream>>>(Vf, stVH, Vt16, 1, B, S, D, H);

  // 5) attention (WMMA QK^T + WMMA PV), streaming, no SxS materialization
  attn_pass_a_kernel<<<ATTN_BLOCKS, 32, 0, stream>>>(Qh16, Kh16, mbuf, lbuf, stP, B, S, D, H);
  attn_pass_b_kernel<<<ATTN_BLOCKS, 32, 0, stream>>>(Qh16, Kh16, Vt16, mbuf, lbuf, stP,
                                                     xf, stX, B, S, D, H);

  // 6) global quant of attention output, final projection (WMMA) -> d_out
  quant_global_kernel<<<EW, 256, 0, stream>>>(xf, NEL, stX, x16);
  gemm_wmma_kernel<<<GEMM_BLOCKS, 256, 0, stream>>>(x16, Wo16, bo, out, nullptr, B * S, D, D);
}
